// GraphLearning_21320217657537
// MI455X (gfx1250) — matrix operand order repaired and checked by reading, not compiled, not run on hardware
//
#include <hip/hip_runtime.h>
#include <hip/hip_bf16.h>
#include <math.h>

// ---------------- CDNA5 WMMA types ----------------
typedef __attribute__((ext_vector_type(16))) __bf16 v16bf;
typedef __attribute__((ext_vector_type(8)))  float  v8f;

#define DDIM 512
#define GNUM 4
#define HDIM 128
#define ROWS_PER_WG 64

#if defined(__has_builtin)
#if __has_builtin(__builtin_amdgcn_sched_group_barrier)
#define SGB(mask, cnt) __builtin_amdgcn_sched_group_barrier((mask), (cnt), 0)
#endif
#endif
#ifndef SGB
#define SGB(mask, cnt)
#endif

__device__ __forceinline__ unsigned short f32_to_bf16_rne(float f) {
    unsigned int x = __float_as_uint(f);
    unsigned int r = x + 0x7FFFu + ((x >> 16) & 1u);
    return (unsigned short)(r >> 16);
}
__device__ __forceinline__ float bf16u_to_f32(unsigned int u16) {
    return __uint_as_float(u16 << 16);
}
// XOR-swizzled LDS offset for the A tile (64 rows x 512 bf16, no padding):
// 8-element chunks permuted by row -> conflict-free b128 lane patterns.
__device__ __forceinline__ int lds_off(int row, int d) {
    return row * DDIM + ((((d >> 3) ^ (row & 15)) << 3) | (d & 7));
}

// ----------------------------------------------------------------------
// Kernel 1: Wbig[d, c] = att[g,d]*W_lin[g,d,h]  (c = g*128+h) in bf16,
// stored pre-swizzled in WMMA-B fragment order (lane slice = 32B contig).
// ----------------------------------------------------------------------
__global__ __launch_bounds__(256)
void k_prep_w(const float* __restrict__ att, const float* __restrict__ W_lin,
              unsigned short* __restrict__ Bpack) {
    int id = blockIdx.x * 256 + threadIdx.x;       // id = k*512 + c
    int k  = id >> 9;
    int c  = id & 511;
    int g  = c >> 7;
    int h  = c & 127;
    float w = att[g * DDIM + k] * W_lin[((size_t)(g * DDIM + k)) * HDIM + h];
    int kt   = k >> 5;
    int kin  = k & 31;
    int ct   = c >> 4;
    int lane = (c & 15) + (kin & 16);
    int kk   = kin & 15;
    size_t idx = ((size_t)(kt * 32 + ct) * 32 + lane) * 16 + kk;
    Bpack[idx] = f32_to_bf16_rne(w);
}

// ----------------------------------------------------------------------
// Kernel 2: v_lr[j,d] = att[g,d] * sum_h W_lin[g,d,h]*w[g,h]
// ----------------------------------------------------------------------
__global__ __launch_bounds__(256)
void k_prep_v(const float* __restrict__ att, const float* __restrict__ W_lin,
              const float* __restrict__ w_al, const float* __restrict__ w_ar,
              float* __restrict__ vlr) {
    int t = blockIdx.x * 256 + threadIdx.x;        // 8 * 512 tasks
    int j = t >> 9;
    int d = t & 511;
    int g = j & 3;
    const float* w = (j < 4) ? (w_al + g * HDIM) : (w_ar + g * HDIM);
    const float* wl = W_lin + ((size_t)(g * DDIM + d)) * HDIM;
    float s = 0.f;
    #pragma unroll 8
    for (int h = 0; h < HDIM; ++h) s = fmaf(wl[h], w[h], s);
    vlr[j * DDIM + d] = att[g * DDIM + d] * s;
}

// ----------------------------------------------------------------------
// Kernel 3: main GEMM.  Block = 16 waves (512 thr), 64 rows x 512 cols.
// Wave w: ALL 64 rows x 32 cols (col tiles 2w, 2w+1) -> 8 acc tiles.
// Each B fragment is reused by 4 WMMAs (4 row tiles) => B L2 traffic
// = N*512*512*2/64 = 0.8 GB total.  B double-buffered (prefetch 1 K-step
// ahead); A single-buffered from swizzled LDS.  sched_group_barrier pins
// DS(A,kt) -> VMEM(B,kt+1) -> 8x WMMA(kt) so loads stay a step ahead.
// ----------------------------------------------------------------------
__global__ __launch_bounds__(512)
void k_gemm(const float* __restrict__ features,
            const unsigned short* __restrict__ Bpack,
            const float* __restrict__ b_lin,
            const float* __restrict__ vlr,
            const float* __restrict__ b_al, const float* __restrict__ b_ar,
            float* __restrict__ hidden, float* __restrict__ alr, int N) {
    __shared__ unsigned short Ash[ROWS_PER_WG * DDIM];   // 65536 B exactly

    const int tid     = threadIdx.x;
    const int rowbase = blockIdx.x * ROWS_PER_WG;

    // ---- stage A tile: 64 rows x 512 f32 -> bf16 swizzled LDS
    const float4* f4 = (const float4*)features;
    #pragma unroll
    for (int i = tid; i < ROWS_PER_WG * 128; i += 512) {
        int row = i >> 7;
        int q   = i & 127;                     // float4 index within row
        int gr  = rowbase + row; if (gr >= N) gr = N - 1;
        float4 v = f4[(size_t)gr * 128 + q];
        ushort4 b;
        b.x = f32_to_bf16_rne(v.x); b.y = f32_to_bf16_rne(v.y);
        b.z = f32_to_bf16_rne(v.z); b.w = f32_to_bf16_rne(v.w);
        *(ushort4*)&Ash[lds_off(row, q * 4)] = b;
    }
    __syncthreads();

    const int wave   = tid >> 5;
    const int lane   = tid & 31;
    const int lane15 = lane & 15;
    const int hb     = (lane >> 4) & 1;        // high-half lane -> K+16 start
    const int wbase  = wave * 32;              // column base (2 col tiles)

    // ---- accumulators seeded with bias (b_lin flat index == output column)
    v8f acc[2][4];
    #pragma unroll
    for (int c = 0; c < 2; ++c) {
        float bv = b_lin[wbase + c * 16 + lane15];
        #pragma unroll
        for (int r = 0; r < 4; ++r)
            #pragma unroll
            for (int e = 0; e < 8; ++e) acc[c][r][e] = bv;
    }

    union Frag { uint4 u[2]; v16bf v; };
    const uint4* bp = (const uint4*)Bpack + (size_t)lane * 2;
    const int ct0 = wave * 2;                  // first of this wave's col tiles

    int rb[4];                                 // per-row-tile LDS base (elems)
    #pragma unroll
    for (int r = 0; r < 4; ++r) rb[r] = (r * 16 + lane15) * DDIM;

    Frag A[4], B[2][2];

#define LOAD_B(kt_, buf_)                                                   \
    do {                                                                    \
        _Pragma("unroll")                                                   \
        for (int c_ = 0; c_ < 2; ++c_) {                                    \
            const size_t fb_ = (size_t)((kt_) * 32 + ct0 + c_) * 64;        \
            B[buf_][c_].u[0] = bp[fb_];                                     \
            B[buf_][c_].u[1] = bp[fb_ + 1];                                 \
        }                                                                   \
    } while (0)
#define LOAD_A(kt_)                                                         \
    do {                                                                    \
        const int c0_ = ((((kt_) * 4 + 2 * hb)     ^ lane15) << 3);         \
        const int c1_ = ((((kt_) * 4 + 2 * hb + 1) ^ lane15) << 3);         \
        _Pragma("unroll")                                                   \
        for (int r_ = 0; r_ < 4; ++r_) {                                    \
            A[r_].u[0] = *(const uint4*)&Ash[rb[r_] + c0_];                 \
            A[r_].u[1] = *(const uint4*)&Ash[rb[r_] + c1_];                 \
        }                                                                   \
    } while (0)

    LOAD_B(0, 0);
    #pragma unroll
    for (int kt = 0; kt < 16; ++kt) {
        LOAD_A(kt);
        if (kt < 15) LOAD_B(kt + 1, (kt + 1) & 1);
        #pragma unroll
        for (int r = 0; r < 4; ++r)
            #pragma unroll
            for (int c = 0; c < 2; ++c)
                acc[c][r] = __builtin_amdgcn_wmma_f32_16x16x32_bf16(
                    false, A[r].v, false, B[kt & 1][c].v, (short)0,
                    acc[c][r], false, false);
        SGB(0x100, 8);                 // 8 DS reads   (A, kt)
        if (kt < 15) SGB(0x020, 4);    // 4 VMEM reads (B, kt+1)
        SGB(0x008, 8);                 // 8 WMMA       (kt)
    }
#undef LOAD_A
#undef LOAD_B

    // ---- write hidden_cat (C/D layout: VGPR e -> rows e / e+8)
    #pragma unroll
    for (int c = 0; c < 2; ++c) {
        const int col = wbase + c * 16 + lane15;
        #pragma unroll
        for (int r = 0; r < 4; ++r) {
            const int rowHalf = rowbase + r * 16 + ((lane & 16) ? 8 : 0);
            #pragma unroll
            for (int e = 0; e < 8; ++e) {
                int grow = rowHalf + e;
                if (grow < N) hidden[(size_t)grow * DDIM + col] = acc[c][r][e];
            }
        }
    }

    // ---- per-node attention scalars from the LDS tile: 64 rows x 8 outputs
    {
        const int row = tid >> 3;
        const int jj  = tid & 7;
        const int grow = rowbase + row;
        if (grow < N) {
            const float* v = vlr + jj * DDIM;
            float s = 0.f;
            #pragma unroll 4
            for (int d = 0; d < DDIM; d += 8) {
                uint4  u  = *(const uint4*)&Ash[lds_off(row, d)];
                float4 w0 = *(const float4*)&v[d];
                float4 w1 = *(const float4*)&v[d + 4];
                s = fmaf(bf16u_to_f32(u.x & 0xffffu), w0.x, s);
                s = fmaf(bf16u_to_f32(u.x >> 16),     w0.y, s);
                s = fmaf(bf16u_to_f32(u.y & 0xffffu), w0.z, s);
                s = fmaf(bf16u_to_f32(u.y >> 16),     w0.w, s);
                s = fmaf(bf16u_to_f32(u.z & 0xffffu), w1.x, s);
                s = fmaf(bf16u_to_f32(u.z >> 16),     w1.y, s);
                s = fmaf(bf16u_to_f32(u.w & 0xffffu), w1.z, s);
                s = fmaf(bf16u_to_f32(u.w >> 16),     w1.w, s);
            }
            const int g = jj & 3;
            const float bias = (jj < 4) ? b_al[g] : b_ar[g];
            alr[(size_t)grow * 8 + jj] = s + bias;
        }
    }
}

// ----------------------------------------------------------------------
// Kernel 4: per-edge sigmoid factors[g,e] = sigmoid(a_l[g,src] + a_r[g,dst])
// ----------------------------------------------------------------------
__global__ __launch_bounds__(256)
void k_edges(const int* __restrict__ src, const int* __restrict__ dst,
             const float* __restrict__ alr, float* __restrict__ factors, int E) {
    int e = blockIdx.x * 256 + threadIdx.x;
    if (e >= E) return;
    int s = src[e], d = dst[e];
    float4 al = *(const float4*)(alr + (size_t)s * 8);       // a_l[0..3]
    float4 ar = *(const float4*)(alr + (size_t)d * 8 + 4);   // a_r[0..3]
    float x;
    x = al.x + ar.x; factors[e]                 = 1.f / (1.f + __expf(-x));
    x = al.y + ar.y; factors[(size_t)E + e]     = 1.f / (1.f + __expf(-x));
    x = al.z + ar.z; factors[(size_t)E * 2 + e] = 1.f / (1.f + __expf(-x));
    x = al.w + ar.w; factors[(size_t)E * 3 + e] = 1.f / (1.f + __expf(-x));
}

// ----------------------------------------------------------------------
extern "C" void kernel_launch(void* const* d_in, const int* in_sizes, int n_in,
                              void* d_out, int out_size, void* d_ws, size_t ws_size,
                              hipStream_t stream) {
    const float* features = (const float*)d_in[0];
    const int*   src      = (const int*)d_in[1];
    const int*   dst      = (const int*)d_in[2];
    const float* att      = (const float*)d_in[3];
    const float* W_lin    = (const float*)d_in[4];
    const float* b_lin    = (const float*)d_in[5];
    const float* w_al     = (const float*)d_in[6];
    const float* b_al     = (const float*)d_in[7];
    const float* w_ar     = (const float*)d_in[8];
    const float* b_ar     = (const float*)d_in[9];

    const int N = in_sizes[0] / DDIM;
    const int E = in_sizes[1];

    char* ws = (char*)d_ws;
    unsigned short* Bpack = (unsigned short*)ws;                     // 512 KB
    float* vlr = (float*)(ws + 512 * 1024);                          // 16 KB
    float* alr = (float*)(ws + 512 * 1024 + 16 * 1024);              // N*8*4 B

    float* hidden  = (float*)d_out;
    float* factors = hidden + (size_t)N * DDIM;

    k_prep_w<<<(DDIM * DDIM) / 256, 256, 0, stream>>>(att, W_lin, Bpack);
    k_prep_v<<<(8 * DDIM) / 256, 256, 0, stream>>>(att, W_lin, w_al, w_ar, vlr);
    k_gemm<<<(N + ROWS_PER_WG - 1) / ROWS_PER_WG, 512, 0, stream>>>(
        features, Bpack, b_lin, vlr, b_al, b_ar, hidden, alr, N);
    k_edges<<<(E + 255) / 256, 256, 0, stream>>>(src, dst, alr, factors, E);
}
